// DNC_44564580663416
// MI455X (gfx1250) — matrix operand order repaired and checked by reading, not compiled, not run on hardware
//
#include <hip/hip_runtime.h>
#include <math.h>

typedef __bf16 bf16;
typedef bf16  v8bf  __attribute__((ext_vector_type(8)));
typedef bf16  v16bf __attribute__((ext_vector_type(16)));
typedef float v8f   __attribute__((ext_vector_type(8)));

#define EPSC 1e-6f

// ---------------- problem sizes ----------------
constexpr int Bb = 16, Tt = 64, INP = 256, Hh = 512, Nn = 256, Wwd = 64, Rr = 4;
constexpr int KG = 1024;          // gates K  = (IN + R*W) + H
constexpr int KI = 512;           // itf K    = H
constexpr int KO = 768;           // out K    = H + R*W
constexpr int NIp = 480;          // itf N padded (471 -> 480)

// ---------------- f32 workspace offsets (in floats) ----------------
constexpr size_t OFF_H   = 0;
constexpr size_t OFF_C   = OFF_H   + (size_t)Bb * Hh;            // 16x512
constexpr size_t OFF_M   = OFF_C   + (size_t)Bb * Hh;            // 16x256x64
constexpr size_t OFF_U   = OFF_M   + (size_t)Bb * Nn * Wwd;      // 16x256
constexpr size_t OFF_L   = OFF_U   + (size_t)Bb * Nn;            // 16x256x256
constexpr size_t OFF_P   = OFF_L   + (size_t)Bb * Nn * Nn;       // 16x256
constexpr size_t OFF_WR  = OFF_P   + (size_t)Bb * Nn;            // 16x4x256
constexpr size_t OFF_WW  = OFF_WR  + (size_t)Bb * Rr * Nn;       // 16x256
constexpr size_t OFF_RW  = OFF_WW  + (size_t)Bb * Nn;            // 16x4x64
constexpr size_t OFF_ITF = OFF_RW  + (size_t)Bb * Rr * Wwd;      // 16x480
constexpr size_t F32_TOTAL = OFF_ITF + (size_t)Bb * NIp;

// ---------------- bf16 workspace offsets (in bf16 elements) ----------------
constexpr size_t BO_WG = 0;                                      // 2048x1024
constexpr size_t BO_WI = BO_WG + (size_t)2048 * 1024;            // 480x512 (transposed W_int)
constexpr size_t BO_WO = BO_WI + (size_t)480 * 512;              // 256x768
constexpr size_t BO_AG = BO_WO + (size_t)256 * 768;              // 16x1024 activations (gates)
constexpr size_t BO_AH = BO_AG + (size_t)16 * 1024;              // 16x768  [h | read_words]

// k_mem dynamic LDS size
constexpr int SMEM_FLOATS = 16384 + 8 * 256 + 4 * 1024 + 3 * 64 + 256 + 32 + 256;
constexpr int SMEM_BYTES  = SMEM_FLOATS * 4;

// ---------------- device helpers ----------------
__device__ __forceinline__ float sigmf(float x) { return 1.f / (1.f + __expf(-x)); }
__device__ __forceinline__ float oneplusf(float x) {
  return 1.f + ((x > 20.f) ? x : log1pf(__expf(x)));
}

// A fragment (16x32 bf16, M = batch rows): ISA 16-bit A layout.
// lane L: m = L&15; k-offset base = (L<16)?0:8; two 16B runs at +0 and +16.
__device__ __forceinline__ v16bf load_fragA(const bf16* __restrict__ A, int row_stride,
                                            int ks, int lane) {
  const int m    = lane & 15;
  const int koff = (lane < 16) ? 0 : 8;
  const bf16* p  = A + (size_t)m * row_stride + ks + koff;
  v8bf lo = *(const v8bf*)(p);
  v8bf hi = *(const v8bf*)(p + 16);
  return __builtin_shufflevector(lo, hi, 0, 1, 2, 3, 4, 5, 6, 7,
                                 8, 9, 10, 11, 12, 13, 14, 15);
}

// B fragment (32x16 bf16, row-major weights W[n][k]): lane holds column n = lane&15,
// k-offset base = (lane<16)?0:8; two 8-element runs at +0 and +16 (matches A).
__device__ __forceinline__ v16bf load_fragB(const bf16* __restrict__ Wrow, int ks, int lane) {
  const int koff = (lane < 16) ? 0 : 8;
  const bf16* p  = Wrow + ks + koff;
  v8bf lo = *(const v8bf*)(p);
  v8bf hi = *(const v8bf*)(p + 16);
  return __builtin_shufflevector(lo, hi, 0, 1, 2, 3, 4, 5, 6, 7,
                                 8, 9, 10, 11, 12, 13, 14, 15);
}

__device__ __forceinline__ float blkmax(float v, float* red, int tid) {
  red[tid] = v; __syncthreads();
  for (int s = 128; s > 0; s >>= 1) {
    if (tid < s) red[tid] = fmaxf(red[tid], red[tid + s]);
    __syncthreads();
  }
  float r = red[0]; __syncthreads();
  return r;
}
__device__ __forceinline__ float blksum(float v, float* red, int tid) {
  red[tid] = v; __syncthreads();
  for (int s = 128; s > 0; s >>= 1) {
    if (tid < s) red[tid] += red[tid + s];
    __syncthreads();
  }
  float r = red[0]; __syncthreads();
  return r;
}

// ---------------- weight conversion ----------------
__global__ void k_conv_gatesw(const float* __restrict__ Wih, const float* __restrict__ Whh,
                              bf16* __restrict__ Wg) {
  size_t i = (size_t)blockIdx.x * blockDim.x + threadIdx.x;
  if (i >= (size_t)2048 * 1024) return;
  int n = (int)(i >> 10), k = (int)(i & 1023);
  float v = (k < 512) ? Wih[(size_t)n * 512 + k] : Whh[(size_t)n * 512 + (k - 512)];
  Wg[i] = (bf16)v;
}
__global__ void k_conv_wint(const float* __restrict__ Wint, bf16* __restrict__ Wt) {
  size_t i = (size_t)blockIdx.x * blockDim.x + threadIdx.x;
  if (i >= (size_t)480 * 512) return;
  int n = (int)(i >> 9), k = (int)(i & 511);
  Wt[i] = (bf16)((n < 471) ? Wint[(size_t)k * 471 + n] : 0.f);
}
__global__ void k_conv_wout(const float* __restrict__ W, bf16* __restrict__ Wo) {
  size_t i = (size_t)blockIdx.x * blockDim.x + threadIdx.x;
  if (i >= (size_t)256 * 768) return;
  Wo[i] = (bf16)W[i];
}

// ---------------- per-step: build bf16 activation A = [x_t | read_words | h] ----------------
__global__ void k_prepA(const float* __restrict__ x, const float* __restrict__ rw,
                        const float* __restrict__ h, bf16* __restrict__ Ag) {
  int i = blockIdx.x * blockDim.x + threadIdx.x;
  if (i >= Bb * KG) return;
  int b = i >> 10, k = i & 1023;
  float v;
  if (k < 256)      v = x[(size_t)b * 256 + k];
  else if (k < 512) v = rw[(size_t)b * 256 + (k - 256)];
  else              v = h[(size_t)b * 512 + (k - 512)];
  Ag[i] = (bf16)v;
}

// ---------------- gates GEMM (WMMA) fused with LSTM pointwise ----------------
// grid 32 (one per 16 columns of H), block 32 (one wave)
__global__ void k_gates(const bf16* __restrict__ Ag, const bf16* __restrict__ Wg,
                        const float* __restrict__ b_ih, const float* __restrict__ b_hh,
                        float* __restrict__ hbuf, float* __restrict__ cbuf,
                        bf16* __restrict__ Ahrw) {
  const int lane = threadIdx.x;
  const int jt   = blockIdx.x;      // 0..31
  v8f acc[4] = {};
  for (int ks = 0; ks < KG; ks += 32) {
    v16bf a = load_fragA(Ag, KG, ks, lane);
#pragma unroll
    for (int q = 0; q < 4; ++q) {
      const int n = q * 512 + jt * 16 + (lane & 15);
      const bf16* wrow = Wg + (size_t)n * KG;
      if (ks + 128 < KG) __builtin_prefetch((const void*)(wrow + ks + 128), 0, 1);
      v16bf bfr = load_fragB(wrow, ks, lane);
      acc[q] = __builtin_amdgcn_wmma_f32_16x16x32_bf16(false, a, false, bfr,
                                                       (short)0, acc[q], false, false);
    }
  }
  const int j = jt * 16 + (lane & 15);
  float bias[4];
#pragma unroll
  for (int q = 0; q < 4; ++q) bias[q] = b_ih[q * 512 + j] + b_hh[q * 512 + j];
  const int mbase = (lane < 16) ? 0 : 8;
#pragma unroll
  for (int v = 0; v < 8; ++v) {
    const int b = mbase + v;
    float gi = acc[0][v] + bias[0], gf = acc[1][v] + bias[1];
    float gg = acc[2][v] + bias[2], go = acc[3][v] + bias[3];
    float cold = cbuf[(size_t)b * Hh + j];
    float cn = sigmf(gf) * cold + sigmf(gi) * tanhf(gg);
    float hn = sigmf(go) * tanhf(cn);
    cbuf[(size_t)b * Hh + j] = cn;
    hbuf[(size_t)b * Hh + j] = hn;
    Ahrw[(size_t)b * KO + j] = (bf16)hn;
  }
}

// ---------------- interface GEMM (WMMA): itf = h @ W_int + b_int ----------------
// grid 30, block 32
__global__ void k_itf(const bf16* __restrict__ Ahrw, const bf16* __restrict__ Wt,
                      const float* __restrict__ b_int, float* __restrict__ itf) {
  const int lane = threadIdx.x;
  const int jt   = blockIdx.x;      // 0..29
  const int n    = jt * 16 + (lane & 15);
  v8f acc = {};
  const bf16* wrow = Wt + (size_t)n * KI;
  for (int ks = 0; ks < KI; ks += 32) {
    v16bf a = load_fragA(Ahrw, KO, ks, lane);
    if (ks + 128 < KI) __builtin_prefetch((const void*)(wrow + ks + 128), 0, 1);
    v16bf bfr = load_fragB(wrow, ks, lane);
    acc = __builtin_amdgcn_wmma_f32_16x16x32_bf16(false, a, false, bfr,
                                                  (short)0, acc, false, false);
  }
  const float bias  = (n < 471) ? b_int[n] : 0.f;
  const int   mbase = (lane < 16) ? 0 : 8;
#pragma unroll
  for (int v = 0; v < 8; ++v) {
    const int b = mbase + v;
    itf[(size_t)b * NIp + n] = acc[v] + bias;
  }
}

// ---------------- output GEMM (WMMA): out = [h|rw] @ W_out.T + b_out ----------------
// grid 16, block 32
__global__ void k_out(const bf16* __restrict__ Ahrw, const bf16* __restrict__ Wo,
                      const float* __restrict__ b_out, float* __restrict__ out) {
  const int lane = threadIdx.x;
  const int n    = blockIdx.x * 16 + (lane & 15);
  v8f acc = {};
  const bf16* wrow = Wo + (size_t)n * KO;
  for (int ks = 0; ks < KO; ks += 32) {
    v16bf a = load_fragA(Ahrw, KO, ks, lane);
    if (ks + 128 < KO) __builtin_prefetch((const void*)(wrow + ks + 128), 0, 1);
    v16bf bfr = load_fragB(wrow, ks, lane);
    acc = __builtin_amdgcn_wmma_f32_16x16x32_bf16(false, a, false, bfr,
                                                  (short)0, acc, false, false);
  }
  const float bias  = b_out[n];
  const int   mbase = (lane < 16) ? 0 : 8;
#pragma unroll
  for (int v = 0; v < 8; ++v) {
    const int b = mbase + v;
    out[(size_t)b * 256 + n] = acc[v] + bias;
  }
}

// ---------------- DNC memory machinery: one block per batch ----------------
// grid 16, block 256, dynamic LDS = SMEM_BYTES
__global__ void k_mem(const float* __restrict__ itf,
                      float* __restrict__ Mg, float* __restrict__ ug,
                      float* __restrict__ Lg, float* __restrict__ pg,
                      float* __restrict__ wrg, float* __restrict__ wwg,
                      float* __restrict__ rwg, bf16* __restrict__ Ahrw) {
  const int b = blockIdx.x, tid = threadIdx.x;
  extern __shared__ float sm[];
  float* Ml    = sm;             // 16384 : M[256][64]
  float* us    = Ml + 16384;     // 256
  float* usrt  = us + 256;       // 256
  float* allo  = usrt + 256;     // 256
  float* cw    = allo + 256;     // 256
  float* wws   = cw + 256;       // 256 (prev ww then new ww)
  float* precs = wws + 256;      // 256
  float* wro   = precs + 256;    // 1024 : wr (old, then new)
  float* fwdv  = wro + 1024;     // 1024
  float* bwdv  = fwdv + 1024;    // 1024
  float* crs   = bwdv + 1024;    // 1024
  float* nrm   = crs + 1024;     // 256
  float* red   = nrm + 256;      // 256
  float* kv    = red + 256;      // 64 : write key
  float* rk    = kv + 64;        // 256 : read keys 4x64
  float* ew    = rk + 256;       // 64 : erase (sigmoided)
  float* wv    = ew + 64;        // 64 : write vector (sigmoided)
  float* sc    = wv + 64;        // 32 : scalars
  int*   rnk   = (int*)(sc + 32);// 256

  const float* it = itf + (size_t)b * NIp;

  // ---- load state + interface slices ----
  for (int i = tid; i < 16384; i += 256) Ml[i] = Mg[(size_t)b * 16384 + i];
  us[tid]    = ug[(size_t)b * 256 + tid];
  precs[tid] = pg[(size_t)b * 256 + tid];
  wws[tid]   = wwg[(size_t)b * 256 + tid];     // previous ww (NW==1)
  rk[tid]    = it[tid];                        // read keys (4x64 = offs [0,256))
  for (int i = tid; i < 1024; i += 256) wro[i] = wrg[(size_t)b * 1024 + i];
  if (tid < 64) {
    kv[tid] = it[260 + tid];
    ew[tid] = sigmf(it[325 + tid]);
    wv[tid] = sigmf(it[389 + tid]);
  }
  if (tid == 0) {
    for (int r = 0; r < 4; ++r) sc[r] = sigmf(it[453 + r]);  // free gates
    sc[4] = sigmf(it[457]);                                  // alloc gate
    sc[5] = sigmf(it[458]);                                  // write gate
    sc[6] = oneplusf(it[324]);                               // write strength
    for (int r = 0; r < 4; ++r) sc[7 + r] = oneplusf(it[256 + r]); // read strengths
    for (int r = 0; r < 4; ++r) {                            // modes softmax (3-way)
      float m0 = it[459 + 3 * r], m1 = it[460 + 3 * r], m2 = it[461 + 3 * r];
      float mx = fmaxf(m0, fmaxf(m1, m2));
      float e0 = __expf(m0 - mx), e1 = __expf(m1 - mx), e2 = __expf(m2 - mx);
      float s = e0 + e1 + e2;
      sc[12 + 3 * r] = e0 / s; sc[13 + 3 * r] = e1 / s; sc[14 + 3 * r] = e2 / s;
    }
  }
  __syncthreads();
  if (tid == 0) {
    float s = 0.f;
    for (int w = 0; w < 64; ++w) s += kv[w] * kv[w];
    sc[24] = sqrtf(s) + EPSC;
  }
  if (tid >= 1 && tid < 5) {
    int r = tid - 1; float s = 0.f;
    for (int w = 0; w < 64; ++w) s += rk[r * 64 + w] * rk[r * 64 + w];
    sc[25 + r] = sqrtf(s) + EPSC;
  }
  __syncthreads();

  // ---- usage update ----
  {
    float psi = 1.f;
    for (int r = 0; r < 4; ++r) psi *= (1.f - sc[r] * wro[r * 256 + tid]);
    float wwa = wws[tid], u = us[tid];
    us[tid] = (u + wwa - u * wwa) * psi;
  }
  __syncthreads();

  // ---- stable-argsort via ranks + allocation weights ----
  {
    float un = us[tid]; int rn = 0;
    for (int j = 0; j < 256; ++j) {
      float uj = us[j];
      rn += (uj < un) || (uj == un && j < tid);
    }
    rnk[tid] = rn; usrt[rn] = un;
  }
  __syncthreads();
  if (tid == 0) {
    float pe = 1.f;
    for (int k = 0; k < 256; ++k) { red[k] = (1.f - usrt[k]) * pe; pe *= usrt[k]; }
  }
  __syncthreads();
  allo[tid] = red[rnk[tid]];
  __syncthreads();

  // ---- write content weights (cosine softmax) ----
  {
    float dot = 0.f, ss = 0.f;
    for (int w = 0; w < 64; ++w) { float m = Ml[tid * 64 + w]; dot += m * kv[w]; ss += m * m; }
    float lg = sc[6] * (dot / ((sqrtf(ss) + EPSC) * sc[24]));
    float mx = blkmax(lg, red, tid);
    float e  = __expf(lg - mx);
    float s  = blksum(e, red, tid);
    cw[tid] = e / s;
  }
  __syncthreads();

  // ---- write weights + sum ----
  wws[tid] = sc[5] * (sc[4] * allo[tid] + (1.f - sc[4]) * cw[tid]);
  __syncthreads();
  {
    float s = blksum(wws[tid], red, tid);
    if (tid == 0) sc[11] = s;
  }
  __syncthreads();

  // ---- memory erase + write ----
  for (int i = tid; i < 16384; i += 256) {
    int n = i >> 6, w = i & 63;
    float wn = wws[n];
    Ml[i] = Ml[i] * (1.f - wn * ew[w]) + wn * wv[w];
  }
  __syncthreads();

  // ---- link update + fwd (row-wise) ----
  {
    float wwi = wws[tid];
    float f0 = 0.f, f1 = 0.f, f2 = 0.f, f3 = 0.f;
    float* Lr = Lg + ((size_t)(b * 256 + tid)) * 256;
    for (int j = 0; j < 256; ++j) {
      float l  = Lr[j];
      float ln = (j == tid) ? 0.f : ((1.f - wwi - wws[j]) * l + wwi * precs[j]);
      Lr[j] = ln;
      f0 += ln * wro[j];        f1 += ln * wro[256 + j];
      f2 += ln * wro[512 + j];  f3 += ln * wro[768 + j];
    }
    fwdv[tid] = f0; fwdv[256 + tid] = f1; fwdv[512 + tid] = f2; fwdv[768 + tid] = f3;
  }
  __threadfence();
  __syncthreads();
  // ---- bwd (column-wise, coalesced over j) ----
  {
    float b0 = 0.f, b1 = 0.f, b2 = 0.f, b3 = 0.f;
    for (int j = 0; j < 256; ++j) {
      float l = Lg[((size_t)(b * 256 + j)) * 256 + tid];
      b0 += l * wro[j];        b1 += l * wro[256 + j];
      b2 += l * wro[512 + j];  b3 += l * wro[768 + j];
    }
    bwdv[tid] = b0; bwdv[256 + tid] = b1; bwdv[512 + tid] = b2; bwdv[768 + tid] = b3;
  }
  __syncthreads();

  // ---- precedence update ----
  precs[tid] = (1.f - sc[11]) * precs[tid] + wws[tid];

  // ---- read content weights ----
  {
    float ss = 0.f;
    for (int w = 0; w < 64; ++w) { float m = Ml[tid * 64 + w]; ss += m * m; }
    nrm[tid] = sqrtf(ss) + EPSC;
  }
  __syncthreads();
  for (int r = 0; r < 4; ++r) {
    float dot = 0.f;
    for (int w = 0; w < 64; ++w) dot += Ml[tid * 64 + w] * rk[r * 64 + w];
    float lg = sc[7 + r] * (dot / (nrm[tid] * sc[25 + r]));
    float mx = blkmax(lg, red, tid);
    float e  = __expf(lg - mx);
    float s  = blksum(e, red, tid);
    crs[r * 256 + tid] = e / s;
  }

  // ---- new read weights ----
  float wrn[4];
#pragma unroll
  for (int r = 0; r < 4; ++r)
    wrn[r] = sc[12 + 3 * r] * bwdv[r * 256 + tid] +
             sc[13 + 3 * r] * fwdv[r * 256 + tid] +
             sc[14 + 3 * r] * crs[r * 256 + tid];
  __syncthreads();
#pragma unroll
  for (int r = 0; r < 4; ++r) wro[r * 256 + tid] = wrn[r];
  __syncthreads();

  // ---- read words: tid -> (r, w), reduce over n ----
  {
    int r = tid >> 6, w = tid & 63;
    float acc = 0.f;
    for (int n = 0; n < 256; ++n) acc += wro[r * 256 + n] * Ml[n * 64 + w];
    rwg[(size_t)b * 256 + tid] = acc;
    Ahrw[(size_t)b * KO + 512 + tid] = (bf16)acc;
  }

  // ---- writeback state ----
  for (int i = tid; i < 16384; i += 256) Mg[(size_t)b * 16384 + i] = Ml[i];
  ug[(size_t)b * 256 + tid]  = us[tid];
  wwg[(size_t)b * 256 + tid] = wws[tid];
  pg[(size_t)b * 256 + tid]  = precs[tid];
  for (int i = tid; i < 1024; i += 256) wrg[(size_t)b * 1024 + i] = wro[i];
}

// ---------------- host launch ----------------
extern "C" void kernel_launch(void* const* d_in, const int* in_sizes, int n_in,
                              void* d_out, int out_size, void* d_ws, size_t ws_size,
                              hipStream_t stream) {
  (void)in_sizes; (void)n_in; (void)out_size; (void)ws_size;

  const float* inputs = (const float*)d_in[0];   // (T,B,IN)
  const float* W_ih   = (const float*)d_in[1];   // (2048,512)
  const float* W_hh   = (const float*)d_in[2];   // (2048,512)
  const float* b_ih   = (const float*)d_in[3];
  const float* b_hh   = (const float*)d_in[4];
  const float* W_int  = (const float*)d_in[5];   // (512,471)
  const float* b_int  = (const float*)d_in[6];
  const float* W_out  = (const float*)d_in[7];   // (256,768)
  const float* b_out  = (const float*)d_in[8];
  float* out = (float*)d_out;                    // (T,B,OUT)

  float* wsf = (float*)d_ws;
  float* h    = wsf + OFF_H;
  float* c    = wsf + OFF_C;
  float* M    = wsf + OFF_M;
  float* usg  = wsf + OFF_U;
  float* link = wsf + OFF_L;
  float* prec = wsf + OFF_P;
  float* wr   = wsf + OFF_WR;
  float* ww   = wsf + OFF_WW;
  float* rw   = wsf + OFF_RW;
  float* itf  = wsf + OFF_ITF;

  bf16* wsb  = (bf16*)(wsf + F32_TOTAL);
  bf16* Wg   = wsb + BO_WG;
  bf16* Wi   = wsb + BO_WI;
  bf16* Wo   = wsb + BO_WO;
  bf16* Ag   = wsb + BO_AG;
  bf16* Ahrw = wsb + BO_AH;

  // zero-init all recurrent state (deterministic every call)
  (void)hipMemsetAsync(wsf, 0, F32_TOTAL * sizeof(float), stream);

  // weight conversion / transpose to bf16
  k_conv_gatesw<<<(2048 * 1024) / 256, 256, 0, stream>>>(W_ih, W_hh, Wg);
  k_conv_wint  <<<(480 * 512) / 256, 256, 0, stream>>>(W_int, Wi);
  k_conv_wout  <<<(256 * 768) / 256, 256, 0, stream>>>(W_out, Wo);

  for (int t = 0; t < Tt; ++t) {
    k_prepA<<<64, 256, 0, stream>>>(inputs + (size_t)t * Bb * INP, rw, h, Ag);
    k_gates<<<32, 32, 0, stream>>>(Ag, Wg, b_ih, b_hh, h, c, Ahrw);
    k_itf  <<<30, 32, 0, stream>>>(Ahrw, Wi, b_int, itf);
    k_mem  <<<16, 256, SMEM_BYTES, stream>>>(itf, M, usg, link, prec, wr, ww, rw, Ahrw);
    k_out  <<<16, 32, 0, stream>>>(Ahrw, Wo, b_out, out + (size_t)t * Bb * 256);
  }
}
